// LM_21345987461650
// MI455X (gfx1250) — matrix operand order repaired and checked by reading, not compiled, not run on hardware
//
#include <hip/hip_runtime.h>
#include <hip/hip_bf16.h>

// ---------------------------------------------------------------------------
// 2-layer LSTM (S=512, B=32, H=D=1024) for gfx1250.
// bf16 WMMA (v_wmma_f32_16x16x32_bf16) with f32 accumulation.
// Weights converted to bf16 once (32MB -> resident in 192MB L2).
// Per-(step,layer) kernel: 32 blocks x 256 threads; each wave owns one
// (gate, 16-col) tile pair and both M tiles of the [32 x 4096] gate matrix.
// All WMMA operands come straight from b128 global loads (row-major layouts
// match the ISA's documented per-lane register layouts exactly).
// ---------------------------------------------------------------------------

#define HDIM 1024
#define BATCH 32
#define SEQ 512
#define NLAYER 2

typedef __attribute__((ext_vector_type(16))) __bf16 bf16x16;
typedef __attribute__((ext_vector_type(8)))  __bf16 bf16x8;
typedef __attribute__((ext_vector_type(8)))  float  f32x8;

union BF16V { bf16x16 v; bf16x8 h[2]; };

static __device__ inline unsigned short f2bf(float f) {
    unsigned u = __builtin_bit_cast(unsigned, f);
    u += 0x7FFFu + ((u >> 16) & 1u);   // round to nearest even
    return (unsigned short)(u >> 16);
}

static __device__ inline bf16x8 ld8(const unsigned short* p) {
    return *reinterpret_cast<const bf16x8*>(p);
}

// ---------------------------------------------------------------- prep kernels

__global__ void k_convert_bf16(const float* __restrict__ src,
                               unsigned short* __restrict__ dst, int n) {
    int i = blockIdx.x * blockDim.x + threadIdx.x;
    if (i < n) dst[i] = f2bf(src[i]);
}

__global__ void k_make_bias(const float* __restrict__ a,
                            const float* __restrict__ b,
                            float* __restrict__ o, int n) {
    int i = blockIdx.x * blockDim.x + threadIdx.x;
    if (i < n) o[i] = a[i] + b[i];
}

// emb[tokens] -> bf16, layout [S][B][H]
__global__ void k_embed_bf16(const int* __restrict__ tokens,
                             const float* __restrict__ emb,
                             unsigned short* __restrict__ xout, int total) {
    int i = blockIdx.x * blockDim.x + threadIdx.x;
    if (i < total) {
        int k  = i & (HDIM - 1);
        int tb = i >> 10;                 // t*B + b
        int tok = tokens[tb];
        xout[i] = f2bf(emb[(size_t)tok * HDIM + k]);
    }
}

// h0/c0 [L][B][H] -> bf16 ping buffer p=0, f32 h, f32 c
__global__ void k_init_state(const float* __restrict__ h0,
                             const float* __restrict__ c0,
                             unsigned short* __restrict__ hbf4, // 4 bufs of 32K
                             float* __restrict__ hf32,
                             float* __restrict__ cf, int total) {
    int i = blockIdx.x * blockDim.x + threadIdx.x;
    if (i < total) {
        int l = i >> 15;                  // 32*1024 = 32768 per layer
        int r = i & 32767;
        hbf4[l * 65536 + r] = f2bf(h0[i]);  // (l, p=0) buffer
        hf32[i] = h0[i];
        cf[i]   = c0[i];
    }
}

__global__ void k_finalize(const float* __restrict__ hf0,
                           const float* __restrict__ hf1,
                           const float* __restrict__ cf0,
                           const float* __restrict__ cf1,
                           float* __restrict__ tail) {
    int i = blockIdx.x * blockDim.x + threadIdx.x;
    if (i < BATCH * HDIM) {
        tail[i]                     = hf0[i];
        tail[BATCH * HDIM + i]      = hf1[i];
        tail[2 * BATCH * HDIM + i]  = cf0[i];
        tail[3 * BATCH * HDIM + i]  = cf1[i];
    }
}

// ------------------------------------------------------------ LSTM step kernel
// gates[32 x 4096] = x @ Wih^T + h @ Whh^T + bias ; then i,f,g,o fusion.
// grid = 32 blocks (32 hidden cols each), block = 256 threads (8 waves).
// wave w: gate = w>>1, N-tile = w&1 (16 cols), both M tiles (rows 0-15,16-31).

__global__ __launch_bounds__(256, 1)
void k_lstm_step(const unsigned short* __restrict__ x,      // [32][1024] bf16
                 const unsigned short* __restrict__ hprev,  // [32][1024] bf16
                 const unsigned short* __restrict__ Wih,    // [4096][1024] bf16
                 const unsigned short* __restrict__ Whh,    // [4096][1024] bf16
                 const float* __restrict__ bias,            // [4096]
                 float* __restrict__ c,                     // [32][1024] in/out
                 unsigned short* __restrict__ hnew_bf,      // [32][1024]
                 float* __restrict__ hnew_f32,              // [32][1024]
                 float* __restrict__ out_f32)               // optional
{
    __shared__ float gl[4][32][33];   // [gate][row][local col], padded

    const int tid  = threadIdx.x;
    const int w    = tid >> 5;
    const int lane = tid & 31;
    const int gate = w >> 1;
    const int nt   = w & 1;
    const int half = lane >> 4;
    const int idx  = lane & 15;
    const int hb   = blockIdx.x;                          // hidden block 0..31
    const int gcol = gate * HDIM + hb * 32 + nt * 16;     // tile base gate col
    const int ncol = gcol + idx;                          // this lane's column

    f32x8 acc0 = {};   // M rows 0-15
    f32x8 acc1 = {};   // M rows 16-31

    const unsigned short* xrow0 = x     + (size_t)idx        * HDIM;
    const unsigned short* xrow1 = x     + (size_t)(16 + idx) * HDIM;
    const unsigned short* hrow0 = hprev + (size_t)idx        * HDIM;
    const unsigned short* hrow1 = hprev + (size_t)(16 + idx) * HDIM;
    const unsigned short* wi    = Wih   + (size_t)ncol       * HDIM;
    const unsigned short* wh    = Whh   + (size_t)ncol       * HDIM;

#pragma unroll 2
    for (int kk = 0; kk < HDIM; kk += 32) {
        BF16V ax0, ax1, ah0, ah1, bi, bh;
        const int aoff = kk + 8 * half;       // A: lanes 0-15 K=0..7/16..23, hi-half +8
        ax0.h[0] = ld8(xrow0 + aoff);  ax0.h[1] = ld8(xrow0 + aoff + 16);
        ax1.h[0] = ld8(xrow1 + aoff);  ax1.h[1] = ld8(xrow1 + aoff + 16);
        ah0.h[0] = ld8(hrow0 + aoff);  ah0.h[1] = ld8(hrow0 + aoff + 16);
        ah1.h[0] = ld8(hrow1 + aoff);  ah1.h[1] = ld8(hrow1 + aoff + 16);
        const int boff = kk + 8 * half;       // B: lanes 0-15 K=0..7/16..23, hi-half +8
        bi.h[0]  = ld8(wi + boff);     bi.h[1]  = ld8(wi + boff + 16);
        bh.h[0]  = ld8(wh + boff);     bh.h[1]  = ld8(wh + boff + 16);

        acc0 = __builtin_amdgcn_wmma_f32_16x16x32_bf16(false, ax0.v, false, bi.v,
                                                       (short)0, acc0, false, false);
        acc1 = __builtin_amdgcn_wmma_f32_16x16x32_bf16(false, ax1.v, false, bi.v,
                                                       (short)0, acc1, false, false);
        acc0 = __builtin_amdgcn_wmma_f32_16x16x32_bf16(false, ah0.v, false, bh.v,
                                                       (short)0, acc0, false, false);
        acc1 = __builtin_amdgcn_wmma_f32_16x16x32_bf16(false, ah1.v, false, bh.v,
                                                       (short)0, acc1, false, false);
    }

    const float bv = bias[ncol];
    const int col = nt * 16 + idx;
#pragma unroll
    for (int v = 0; v < 8; ++v) {          // C/D layout: lane idx = N, M = v + 8*half
        gl[gate][v + 8 * half][col]      = acc0[v] + bv;
        gl[gate][16 + v + 8 * half][col] = acc1[v] + bv;
    }
    __syncthreads();

    for (int e = tid; e < BATCH * 32; e += 256) {
        const int b = e >> 5;
        const int j = e & 31;
        const int J = hb * 32 + j;
        const float iv = 1.f / (1.f + __expf(-gl[0][b][j]));
        const float fv = 1.f / (1.f + __expf(-gl[1][b][j]));
        const float gv = tanhf(gl[2][b][j]);
        const float ov = 1.f / (1.f + __expf(-gl[3][b][j]));
        const float co = c[b * HDIM + J];
        const float cn = fv * co + iv * gv;
        const float hn = ov * tanhf(cn);
        c[b * HDIM + J]        = cn;
        hnew_bf[b * HDIM + J]  = f2bf(hn);
        hnew_f32[b * HDIM + J] = hn;
        if (out_f32) out_f32[b * HDIM + J] = hn;
    }
}

// ------------------------------------------------------------------ host side

extern "C" void kernel_launch(void* const* d_in, const int* in_sizes, int n_in,
                              void* d_out, int out_size, void* d_ws, size_t ws_size,
                              hipStream_t stream) {
    const int*   tokens = (const int*)d_in[0];
    const float* h0     = (const float*)d_in[1];
    const float* c0     = (const float*)d_in[2];
    const float* emb    = (const float*)d_in[3];
    const float* Wih0   = (const float*)d_in[4];
    const float* Whh0   = (const float*)d_in[5];
    const float* bih0   = (const float*)d_in[6];
    const float* bhh0   = (const float*)d_in[7];
    const float* Wih1   = (const float*)d_in[8];
    const float* Whh1   = (const float*)d_in[9];
    const float* bih1   = (const float*)d_in[10];
    const float* bhh1   = (const float*)d_in[11];
    float* out = (float*)d_out;

    char* ws = (char*)d_ws;
    const size_t SZ_W   = (size_t)4 * HDIM * HDIM * 2;    // 8 MB per bf16 matrix
    const size_t OFF_W  = 0;                              // 4 matrices: 32 MB
    const size_t OFF_B0 = 4 * SZ_W;
    const size_t OFF_B1 = OFF_B0 + 4 * HDIM * 4;
    const size_t OFF_X  = OFF_B1 + 4 * HDIM * 4;          // 33587200 (256-aligned)
    const size_t SZ_X   = (size_t)SEQ * BATCH * HDIM * 2; // 32 MB
    const size_t OFF_HB = OFF_X + SZ_X;                   // 4 bf16 h bufs (l,p)
    const size_t OFF_HF = OFF_HB + (size_t)4 * BATCH * HDIM * 2;
    const size_t OFF_C  = OFF_HF + (size_t)NLAYER * BATCH * HDIM * 4;

    unsigned short* wb[4] = {
        (unsigned short*)(ws + OFF_W + 0 * SZ_W),   // Wih0
        (unsigned short*)(ws + OFF_W + 1 * SZ_W),   // Whh0
        (unsigned short*)(ws + OFF_W + 2 * SZ_W),   // Wih1
        (unsigned short*)(ws + OFF_W + 3 * SZ_W) }; // Whh1
    float* bias0 = (float*)(ws + OFF_B0);
    float* bias1 = (float*)(ws + OFF_B1);
    unsigned short* xb   = (unsigned short*)(ws + OFF_X);
    unsigned short* hbf  = (unsigned short*)(ws + OFF_HB);
    float* hf32 = (float*)(ws + OFF_HF);
    float* cf   = (float*)(ws + OFF_C);

    // --- prep: weights -> bf16, biases fused, embedding gather -> bf16, state
    const int nW = 4 * HDIM * HDIM;
    k_convert_bf16<<<(nW + 255) / 256, 256, 0, stream>>>(Wih0, wb[0], nW);
    k_convert_bf16<<<(nW + 255) / 256, 256, 0, stream>>>(Whh0, wb[1], nW);
    k_convert_bf16<<<(nW + 255) / 256, 256, 0, stream>>>(Wih1, wb[2], nW);
    k_convert_bf16<<<(nW + 255) / 256, 256, 0, stream>>>(Whh1, wb[3], nW);
    k_make_bias<<<(4 * HDIM + 255) / 256, 256, 0, stream>>>(bih0, bhh0, bias0, 4 * HDIM);
    k_make_bias<<<(4 * HDIM + 255) / 256, 256, 0, stream>>>(bih1, bhh1, bias1, 4 * HDIM);
    const int nX = SEQ * BATCH * HDIM;
    k_embed_bf16<<<(nX + 255) / 256, 256, 0, stream>>>(tokens, emb, xb, nX);
    const int nS = NLAYER * BATCH * HDIM;
    k_init_state<<<(nS + 255) / 256, 256, 0, stream>>>(h0, c0, hbf, hf32, cf, nS);

    // --- recurrence: 2 launches per step (layer1 consumes layer0's fresh h)
    const size_t HS = (size_t)BATCH * HDIM;   // 32768 elements
    for (int t = 0; t < SEQ; ++t) {
        const int rp = t & 1, wp = (t + 1) & 1;
        // layer 0
        k_lstm_step<<<32, 256, 0, stream>>>(
            xb + (size_t)t * HS,
            hbf + (0 * 2 + rp) * HS,
            wb[0], wb[1], bias0,
            cf + 0 * HS,
            hbf + (0 * 2 + wp) * HS,
            hf32 + 0 * HS,
            (float*)nullptr);
        // layer 1 (x = layer0's new h, bf16)
        k_lstm_step<<<32, 256, 0, stream>>>(
            hbf + (0 * 2 + wp) * HS,
            hbf + (1 * 2 + rp) * HS,
            wb[2], wb[3], bias1,
            cf + 1 * HS,
            hbf + (1 * 2 + wp) * HS,
            hf32 + 1 * HS,
            out + (size_t)t * HS);
    }

    // --- tail: hF (both layers) then cF (both layers)
    k_finalize<<<(BATCH * HDIM + 255) / 256, 256, 0, stream>>>(
        hf32 + 0 * HS, hf32 + 1 * HS, cf + 0 * HS, cf + 1 * HS,
        out + (size_t)SEQ * HS);
}
